// WhiteboxGATCRATELayer_85177791414404
// MI455X (gfx1250) — matrix operand order repaired and checked by reading, not compiled, not run on hardware
//
#include <hip/hip_runtime.h>

#define NN 4096
#define DD 512
#define KH 8
#define RR 64

typedef __attribute__((ext_vector_type(16))) _Float16 v16h;
typedef __attribute__((ext_vector_type(8)))  _Float16 v8h;
typedef __attribute__((ext_vector_type(8)))  float    v8f;
typedef __attribute__((ext_vector_type(2)))  float    v2f;
typedef __attribute__((ext_vector_type(4)))  unsigned int v4u;
typedef __attribute__((ext_vector_type(8)))  int      v8i;
typedef __attribute__((ext_vector_type(4)))  int      v4i;

union V16U { v16h v; v8h h[2]; };

// A fragment (16x32 f16): lane holds row = lane%16; K runs [kb+8*hi .. +7] and [kb+16+8*hi .. +7]
__device__ __forceinline__ v16h ldA(const _Float16* row, int kb, int hi) {
  V16U u;
  const _Float16* p = row + kb + 8 * hi;
  u.h[0] = *(const v8h*)p;
  u.h[1] = *(const v8h*)(p + 16);
  return u.v;
}
// B fragment (32x16 f16): lane holds col = lane%16; K runs [kb+8*hi .. +7] and [kb+16+8*hi .. +7]
__device__ __forceinline__ v16h ldB(const _Float16* col, int kb, int hi) {
  V16U u;
  const _Float16* p = col + kb + 8 * hi;
  u.h[0] = *(const v8h*)p;
  u.h[1] = *(const v8h*)(p + 16);
  return u.v;
}
__device__ __forceinline__ v8f wmma16(v16h a, v16h b, v8f c) {
  return __builtin_amdgcn_wmma_f32_16x16x32_f16(false, a, false, b, (short)0, c, false, false);
}

// ---------------- conversion / transposition ----------------
__global__ void k_convert(const float* __restrict__ H, const float* __restrict__ U,
                          _Float16* __restrict__ Hf, _Float16* __restrict__ Ut,
                          _Float16* __restrict__ WT, float* __restrict__ out) {
  size_t i = (size_t)blockIdx.x * blockDim.x + threadIdx.x;
  if (i == 0) out[(size_t)NN * DD] = 0.0f;   // zero the orth-loss accumulator
  if (i < (size_t)NN * DD) Hf[i] = (_Float16)H[i];
  if (i < (size_t)KH * DD * RR) {
    int k = (int)(i / (DD * RR));
    int rem = (int)(i - (size_t)k * DD * RR);
    int d = rem / RR, r = rem - d * RR;
    float v = U[i];
    Ut[((size_t)k * RR + r) * DD + d] = (_Float16)v;
    WT[(size_t)d * DD + k * RR + r]   = (_Float16)v;
  }
}

// ---------------- projection: Hk = H @ U_k, plus transposed copy ----------------
__global__ __launch_bounds__(128) void k_proj(const _Float16* __restrict__ Hf,
                                              const _Float16* __restrict__ Ut,
                                              _Float16* __restrict__ Hk,
                                              _Float16* __restrict__ HkT) {
  int lane = threadIdx.x & 31, wid = threadIdx.x >> 5;
  int g = blockIdx.x * 4 + wid;          // 2048 waves: 8 heads x 256 row tiles
  int head = g >> 8, i0 = (g & 255) << 4;
  int hi = lane >> 4, lo = lane & 15;
  const _Float16* Utk  = Ut + (size_t)head * RR * DD;
  const _Float16* arow = Hf + (size_t)(i0 + lo) * DD;
  v8f acc[4] = {};
  for (int d0 = 0; d0 < DD; d0 += 32) {
    v16h a = ldA(arow, d0, hi);
#pragma unroll
    for (int t = 0; t < 4; ++t) {
      v16h b = ldB(Utk + (size_t)(t * 16 + lo) * DD, d0, hi);
      acc[t] = wmma16(a, b, acc[t]);
    }
  }
  _Float16* Hk_h  = Hk  + (size_t)head * NN * RR;
  _Float16* HkT_h = HkT + (size_t)head * RR * NN;
#pragma unroll
  for (int t = 0; t < 4; ++t)
#pragma unroll
    for (int v = 0; v < 8; ++v) {
      int n = i0 + v + 8 * hi, r = t * 16 + lo;
      _Float16 x = (_Float16)acc[t][v];
      Hk_h[(size_t)n * RR + r]  = x;
      HkT_h[(size_t)r * NN + n] = x;
    }
}

// ---------------- fused masked flash attention per (head, 16-row tile) ----------------
// Adjacency tiles are fetched by the Tensor Data Mover: one tensor_load_to_lds per
// 16x32 f32 tile (2 KB) per wave per chunk, tracked with TENSORcnt.
__global__ __launch_bounds__(128) void k_attn(const _Float16* __restrict__ Hk,
                                              const _Float16* __restrict__ HkT,
                                              const float* __restrict__ adj,
                                              _Float16* __restrict__ Z) {
  __shared__ __align__(16) float    mlds[4][16 * 32];
  __shared__ __align__(16) _Float16 wlds[4][16 * 32];
  int lane = threadIdx.x & 31, wid = threadIdx.x >> 5;
  int g = blockIdx.x * 4 + wid;          // 2048 waves
  int head = g >> 8, i0 = (g & 255) << 4;
  int hi = lane >> 4, lo = lane & 15;
  const _Float16* Hk_h  = Hk  + (size_t)head * NN * RR;
  const _Float16* HkT_h = HkT + (size_t)head * RR * NN;
  float*    mw = mlds[wid];
  _Float16* ww = wlds[wid];

  // ---- loop-invariant part of the TDM descriptor (D#) ----
  // flat->LDS mapping keeps the LDS byte offset in addr[31:0]
  unsigned lds_addr = (unsigned)(unsigned long long)(uintptr_t)mw;
  v8i g1;
  g1[0] = 0x20000;                 // data_size = 2 (4 bytes); no multicast/pad/iterate
  g1[1] = (NN & 0xFFFF) << 16;     // tensor_dim0[15:0] = 4096
  g1[2] = (NN & 0xFFFF) << 16;     // tensor_dim0[31:16]=0 | tensor_dim1[15:0] = 4096
  g1[3] = 32 << 16;                // tensor_dim1[31:16]=0 | tile_dim0 = 32
  g1[4] = 16;                      // tile_dim1 = 16, tile_dim2 = 0
  g1[5] = NN;                      // tensor_dim0_stride[31:0] = 4096
  g1[6] = 0;                       // stride[47:32]=0 | tensor_dim1_stride[15:0]=0
  g1[7] = 0;
  v4i g2 = {0, 0, 0, 0};           // 2-D tensor: groups 2/3 unused (zero)
  v4i g3 = {0, 0, 0, 0};
  v8i g4 = {0, 0, 0, 0, 0, 0, 0, 0};  // extended descriptor words (unused, 2-D)

  // Q fragments for this 16-row tile (K-dim r = 64 -> two 32-chunks), loaded once
  const _Float16* qrow = Hk_h + (size_t)(i0 + lo) * RR;
  v16h qa0 = ldA(qrow, 0, hi);
  v16h qa1 = ldA(qrow, 32, hi);

  v8f o[4] = {};
  float mrun[8], lrun[8];
#pragma unroll
  for (int v = 0; v < 8; ++v) { mrun[v] = -1e30f; lrun[v] = 0.0f; }
  const float scale = 0.125f;            // 1/sqrt(64)

  for (int j0 = 0; j0 < NN; j0 += 32) {
    // ---- issue TDM load of adj tile [16 rows][32 cols] (f32) into per-wave LDS ----
    {
      unsigned long long ga = (unsigned long long)(uintptr_t)
          (adj + (size_t)i0 * NN + j0);  // global byte address of tile start
      v4u g0;
      g0[0] = 1u;                                         // count=1, user mode
      g0[1] = lds_addr;                                   // LDS destination
      g0[2] = (unsigned)(ga & 0xFFFFFFFFu);               // global_addr[31:0]
      g0[3] = (unsigned)((ga >> 32) & 0x01FFFFFFu)        // global_addr[56:32]
              | (2u << 30);                               // type = 2 (image/tensor)
      __builtin_amdgcn_tensor_load_to_lds(g0, g1, g2, g3, g4, 0);
    }

    // scores S = Q K^T for two 16-col key tiles, K-dim r = 64 (overlaps TDM)
    v8f s0 = {}, s1 = {};
    {
      const _Float16* kc0 = Hk_h + (size_t)(j0 + lo) * RR;
      const _Float16* kc1 = Hk_h + (size_t)(j0 + 16 + lo) * RR;
      s0 = wmma16(qa0, ldB(kc0, 0, hi), s0);
      s0 = wmma16(qa1, ldB(kc0, 32, hi), s0);
      s1 = wmma16(qa0, ldB(kc1, 0, hi), s1);
      s1 = wmma16(qa1, ldB(kc1, 32, hi), s1);
    }
    __builtin_amdgcn_s_wait_tensorcnt(0);  // adj tile resident in LDS
    __syncthreads();

    // online softmax: row max over unmasked scores is a valid shift (self-loop rows)
    float corr[8];
#pragma unroll
    for (int v = 0; v < 8; ++v) {
      float a = fmaxf(s0[v], s1[v]);
#pragma unroll
      for (int m = 1; m < 16; m <<= 1) a = fmaxf(a, __shfl_xor(a, m, 32));
      a *= scale;
      float nm = fmaxf(mrun[v], a);
      corr[v] = __expf(mrun[v] - nm);
      mrun[v] = nm;
    }
#pragma unroll
    for (int v = 0; v < 8; ++v) {
      int row = v + 8 * hi;              // C-layout row of this VGPR
      float p0 = mw[row * 32 + lo]      * __expf(s0[v] * scale - mrun[v]);
      float p1 = mw[row * 32 + 16 + lo] * __expf(s1[v] * scale - mrun[v]);
      float cs = p0 + p1;
#pragma unroll
      for (int m = 1; m < 16; m <<= 1) cs += __shfl_xor(cs, m, 32);
      lrun[v] = lrun[v] * corr[v] + cs;
      ww[row * 32 + lo]      = (_Float16)p0;    // C-layout -> LDS for A-layout reload
      ww[row * 32 + 16 + lo] = (_Float16)p1;
#pragma unroll
      for (int t = 0; t < 4; ++t) o[t][v] *= corr[v];
    }
    __syncthreads();

    // O += P @ V  (V = Hk rows j0..j0+31, via transposed copy -> contiguous B runs)
    v16h wa = ldA(ww + lo * 32, 0, hi);
#pragma unroll
    for (int t = 0; t < 4; ++t) {
      const _Float16* vc = HkT_h + (size_t)(t * 16 + lo) * NN + j0;
      o[t] = wmma16(wa, ldB(vc, 0, hi), o[t]);
    }
  }

  // normalize and emit Z_agg in [n][k*64+r] layout (A matrix of the output GEMM)
  int cb = head * RR;
#pragma unroll
  for (int v = 0; v < 8; ++v) {
    float inv = 1.0f / lrun[v];
    int n = i0 + v + 8 * hi;
#pragma unroll
    for (int t = 0; t < 4; ++t)
      Z[(size_t)n * (KH * RR) + cb + t * 16 + lo] = (_Float16)(o[t][v] * inv);
  }
}

// ---------------- output GEMM: relu(H + 0.5 * Zagg @ Ucat^T - thr) ----------------
__global__ __launch_bounds__(128) void k_out(const _Float16* __restrict__ Z,
                                             const _Float16* __restrict__ WT,
                                             const float* __restrict__ H,
                                             const float* __restrict__ thr,
                                             float* __restrict__ out) {
  int lane = threadIdx.x & 31, wid = threadIdx.x >> 5;
  int g = blockIdx.x * 4 + wid;          // 2048 waves: 256 row tiles x 8 col groups
  int i0 = (g >> 3) << 4, d0 = (g & 7) << 6;
  int hi = lane >> 4, lo = lane & 15;
  const _Float16* arow = Z + (size_t)(i0 + lo) * DD;
  v8f acc[4] = {};
  for (int kb = 0; kb < DD; kb += 32) {
    v16h a = ldA(arow, kb, hi);
#pragma unroll
    for (int t = 0; t < 4; ++t) {
      v16h b = ldB(WT + (size_t)(d0 + t * 16 + lo) * DD, kb, hi);
      acc[t] = wmma16(a, b, acc[t]);
    }
  }
#pragma unroll
  for (int t = 0; t < 4; ++t) {
    int d = d0 + t * 16 + lo;
    float th = thr[d];
#pragma unroll
    for (int v = 0; v < 8; ++v) {
      int n = i0 + v + 8 * hi;
      float x = H[(size_t)n * DD + d] + 0.5f * acc[t][v] - th;
      out[(size_t)n * DD + d] = fmaxf(x, 0.0f);
    }
  }
}

// ---------------- orthogonality loss with f32 WMMA (16x16x4) ----------------
__global__ __launch_bounds__(128) void k_orth(const float* __restrict__ U,
                                              float* __restrict__ loss) {
  int lane = threadIdx.x & 31, wid = threadIdx.x >> 5;
  int hi = lane >> 4, lo = lane & 15;
  int p = blockIdx.x;                    // 28 blocks, one per (k<l) pair
  int k = 0, rem = p;
  while (rem >= 7 - k) { rem -= 7 - k; ++k; }
  int l = k + 1 + rem;
  const float* Uk = U + (size_t)k * DD * RR;
  const float* Ul = U + (size_t)l * DD * RR;
  float part = 0.0f;
#pragma unroll
  for (int tt = 0; tt < 4; ++tt) {       // each wave owns 4 of the 16 (rt,st) tiles
    int t = wid * 4 + tt;
    int rt = (t >> 2) * 16, st = (t & 3) * 16;
    v8f c = {};
    for (int d0 = 0; d0 < DD; d0 += 4) {
      int dA = d0 + 2 * hi;              // f32 A/B layout: VGPR v -> K = v + 2*(lane/16)
      v2f a, b;
      a[0] = Uk[(size_t)(dA + 0) * RR + rt + lo];
      a[1] = Uk[(size_t)(dA + 1) * RR + rt + lo];
      b[0] = Ul[(size_t)(dA + 0) * RR + st + lo];
      b[1] = Ul[(size_t)(dA + 1) * RR + st + lo];
      c = __builtin_amdgcn_wmma_f32_16x16x4_f32(false, a, false, b, (short)0, c, false, false);
    }
#pragma unroll
    for (int v = 0; v < 8; ++v) part += c[v] * c[v];
  }
#pragma unroll
  for (int m = 1; m < 32; m <<= 1) part += __shfl_xor(part, m, 32);
  if (lane == 0) atomicAdd(loss, part);
}

extern "C" void kernel_launch(void* const* d_in, const int* in_sizes, int n_in,
                              void* d_out, int out_size, void* d_ws, size_t ws_size,
                              hipStream_t stream) {
  (void)in_sizes; (void)n_in; (void)out_size; (void)ws_size;
  const float* H   = (const float*)d_in[0];
  const float* adj = (const float*)d_in[1];
  const float* U   = (const float*)d_in[2];
  const float* thr = (const float*)d_in[3];
  float* out = (float*)d_out;

  // workspace layout (~17 MB of f16)
  _Float16* Hf  = (_Float16*)d_ws;                 // [N][D]
  _Float16* Ut  = Hf  + (size_t)NN * DD;           // [K][R][D]
  _Float16* WT  = Ut  + (size_t)KH * RR * DD;      // [D][K*R]
  _Float16* Hk  = WT  + (size_t)DD * DD;           // [K][N][R]
  _Float16* HkT = Hk  + (size_t)KH * NN * RR;      // [K][R][N]
  _Float16* Z   = HkT + (size_t)KH * NN * RR;      // [N][K*R]

  k_convert<<<(NN * DD) / 256, 256, 0, stream>>>(H, U, Hf, Ut, WT, out);
  k_proj<<<512, 128, 0, stream>>>(Hf, Ut, Hk, HkT);
  k_attn<<<512, 128, 0, stream>>>(Hk, HkT, adj, Z);
  k_out<<<512, 128, 0, stream>>>(Z, WT, H, thr, out);
  k_orth<<<28, 128, 0, stream>>>(U, out + (size_t)NN * DD);
}
